// GNN_70970039599960
// MI455X (gfx1250) — matrix operand order repaired and checked by reading, not compiled, not run on hardware
//
#include <hip/hip_runtime.h>
#include <hip/hip_bf16.h>
#include <float.h>

typedef __attribute__((ext_vector_type(16))) _Float16 v16h;
typedef __attribute__((ext_vector_type(8)))  float    v8f;

#define WPB 4  // waves per block for WMMA kernels

// ---------- order-preserving float <-> uint encoding for atomic max/min ----------
__device__ __forceinline__ unsigned fenc(float f) {
  unsigned b = __float_as_uint(f);
  return (b & 0x80000000u) ? ~b : (b | 0x80000000u);
}
__device__ __forceinline__ float fdec(unsigned u) {
  unsigned b = (u & 0x80000000u) ? (u & 0x7FFFFFFFu) : ~u;
  return __uint_as_float(b);
}

// Branch-free tanh: 1 - 2/(exp(2x)+1). Saturates to +/-1 for |x| large,
// monotone, no exec-mask branching (v_exp_f32 + v_rcp_f32). f16-grade accuracy.
__device__ __forceinline__ float fast_tanh(float x) {
  float e = __expf(2.0f * x);
  float r = __builtin_amdgcn_rcpf(e + 1.0f);
  return fmaf(-2.0f, r, 1.0f);
}

// ---------- weight pack: f32 W[K][N] -> f16 WMMA B fragments ----------
// Fragment (kt,nt): 32 lanes x 16 halves. lane l: N = nt*16 + (l&15),
// half h: K = kt*32 + (l>>4)*8 + (h&7) + (h>>3)*16   (ISA 7.12.2 B layout, wave32)
__global__ void k_pack_w(const float* __restrict__ W, _Float16* __restrict__ dst,
                         int K, int N) {
  int idx = blockIdx.x * blockDim.x + threadIdx.x;
  if (idx >= K * N) return;
  int h    = idx & 15;
  int lane = (idx >> 4) & 31;
  int f    = idx >> 9;
  int NT = N >> 4;
  int kt = f / NT, nt = f - kt * NT;
  int Kr = kt * 32 + ((lane >> 4) << 3) + (h & 7) + ((h >> 3) << 4);
  int Nc = nt * 16 + (lane & 15);
  dst[idx] = (_Float16)W[Kr * N + Nc];
}

// ---------- init kernels ----------
__global__ void k_init_wstats(float* wsum, unsigned* wmax, unsigned* wmin, int* cnt, int n) {
  int idx = blockIdx.x * blockDim.x + threadIdx.x;
  if (idx < n * 4) {
    wsum[idx] = 0.f;
    wmax[idx] = fenc(-FLT_MAX);
    wmin[idx] = fenc( FLT_MAX);
  }
  if (idx < n) cnt[idx] = 0;
}

__global__ void k_init_mstats(float* msum, unsigned* mmax, unsigned* mmin, int total) {
  int idx = blockIdx.x * blockDim.x + threadIdx.x;
  if (idx >= total) return;
  msum[idx] = 0.f;
  mmax[idx] = fenc(-FLT_MAX);
  mmin[idx] = fenc( FLT_MAX);
}

// ---------- edge-weight stats (prepare stage) ----------
__global__ void k_edge_wstats(const float* __restrict__ ew, const int* __restrict__ edges, int E,
                              float* wsum, unsigned* wmax, unsigned* wmin, int* cnt) {
  int idx = blockIdx.x * blockDim.x + threadIdx.x;
  if (idx >= E * 4) return;
  int e = idx >> 2, c = idx & 3;
  int node = edges[e];
  float w = ew[e * 4 + c];
  atomicAdd(&wsum[node * 4 + c], w);
  atomicMax(&wmax[node * 4 + c], fenc(w));
  atomicMin(&wmin[node * 4 + c], fenc(w));
  if (c == 0) atomicAdd(&cnt[node], 1);
}

// ---------- build initial reps [n,32] = [x(8), mean(4), max(4), min(4), sum(4), 0(8)] ----------
__global__ void k_prepare(const float* __restrict__ x, const float* __restrict__ wsum,
                          const unsigned* __restrict__ wmax, const unsigned* __restrict__ wmin,
                          const int* __restrict__ cnt, float* __restrict__ reps, int n) {
  int idx = blockIdx.x * blockDim.x + threadIdx.x;
  if (idx >= n * 32) return;
  int node = idx >> 5, k = idx & 31;
  float v = 0.f;
  if (k < 8)       v = x[node * 8 + k];
  else if (k < 12) v = wsum[node * 4 + (k - 8)] / fmaxf((float)cnt[node], 1.f);
  else if (k < 16) v = fdec(wmax[node * 4 + (k - 12)]);
  else if (k < 20) v = fdec(wmin[node * 4 + (k - 16)]);
  else if (k < 24) v = wsum[node * 4 + (k - 20)];
  reps[idx] = v;
}

// ---------- edge message scatter: msgs = reps[nbr], reduce into dest node ----------
// 32 lanes per edge -> edge index is wave-uniform: scalarize node/nbr so the
// gather and atomic streams use scalar-base + lane-offset addressing.
__global__ void k_edge_scatter(const float* __restrict__ reps, const int* __restrict__ edges, int E,
                               float* msum, unsigned* mmax, unsigned* mmin) {
  int idx = blockIdx.x * blockDim.x + threadIdx.x;
  if (idx >= E * 32) return;
  int e = idx >> 5, c = idx & 31;           // e uniform across the wave32
  int node = __builtin_amdgcn_readfirstlane(edges[e]);
  int nbr  = __builtin_amdgcn_readfirstlane(edges[E + e]);
  float v = reps[nbr * 32 + c];             // one coalesced 128B line per edge
  atomicAdd(&msum[node * 32 + c], v);
  atomicMax(&mmax[node * 32 + c], fenc(v));
  atomicMin(&mmin[node * 32 + c], fenc(v));
}

// ---------- per-wave WMMA GEMM + tanh: [16,K] x W[K,N] + bias ----------
// A tile in LDS as f16 [16][K]; B pre-packed in fragment layout; output tanh()
// written to LDS as f16 [16][N] (Olds) and/or f32 [16][N] (Of32).
__device__ __forceinline__ void wave_gemm(const _Float16* Alds, int K,
                                          const _Float16* __restrict__ Bpk,
                                          const float* __restrict__ bias, int N,
                                          _Float16* Olds, float* Of32, int lane) {
  const int NT = N >> 4, KT = K >> 5;
  const int row   = lane & 15;        // C/D: N column; A: M row
  const int kgrp  = (lane >> 4) << 3; // A K sub-group base (0 or 8)
  const int mbase = (lane >> 4) << 3; // C/D M base (0 or 8)
  for (int nt = 0; nt < NT; ++nt) {
    // prefetch next N-tile's first B fragment (global_prefetch_b8)
    if (nt + 1 < NT)
      __builtin_prefetch(Bpk + (size_t)((nt + 1) * 32 + lane) * 16, 0, 3);
    float bv = bias[nt * 16 + row];
    v8f c;
#pragma unroll
    for (int v = 0; v < 8; ++v) c[v] = bv;
    for (int kt = 0; kt < KT; ++kt) {
      // A fragment (16x32 f16): lane holds M=row, K = kt*32 + kgrp + {0..7, 16..23}
      const _Float16* ap = Alds + row * K + kt * 32 + kgrp;
      v16h a;
#pragma unroll
      for (int i = 0; i < 8; ++i) { a[i] = ap[i]; a[i + 8] = ap[16 + i]; }
      // B fragment: 32 bytes per lane, contiguous
      v16h b = ((const v16h*)Bpk)[(kt * NT + nt) * 32 + lane];
      c = __builtin_amdgcn_wmma_f32_16x16x32_f16(false, a, false, b, (short)0, c,
                                                 false, false);
    }
#pragma unroll
    for (int v = 0; v < 8; ++v) {
      float t = fast_tanh(c[v]);
      int m = mbase + v;
      int ncol = nt * 16 + row;
      if (Olds) Olds[m * N + ncol] = (_Float16)t;
      if (Of32) Of32[m * N + ncol] = t;
    }
  }
}

// ---------- node update: h=[reps,mean,max,min] (128) -> 256 -> 128 -> 32, l2-normalize ----------
__global__ __launch_bounds__(32 * WPB) void k_update(
    float* __restrict__ reps, const float* __restrict__ msum,
    const unsigned* __restrict__ mmax, const unsigned* __restrict__ mmin,
    const int* __restrict__ cnt,
    const _Float16* __restrict__ pW1, const float* __restrict__ b1,
    const _Float16* __restrict__ pW2, const float* __restrict__ b2,
    const _Float16* __restrict__ pW3, const float* __restrict__ b3, int n) {
  __shared__ __align__(16) _Float16 smem[WPB][16 * 128 + 16 * 256]; // 12KB/wave
  int wave = threadIdx.x >> 5, lane = threadIdx.x & 31;
  int tile = blockIdx.x * WPB + wave;
  int ntiles = (n + 15) >> 4;
  bool active = tile < ntiles;
  int t0 = active ? tile * 16 : 0;

  _Float16* A  = &smem[wave][0];        // [16][128]
  _Float16* B  = &smem[wave][16 * 128]; // [16][256]
  float*    Bf = (float*)B;             // [16][32] layer-3 f32 output (reuses B)
  float*    scale = (float*)A;          // [16] (reuses A after GEMM3)

  // build h tile [16][128] in f16
  for (int t = lane; t < 16 * 128; t += 32) {
    int m = t >> 7, k = t & 127, c = k & 31;
    int node = t0 + m; if (node >= n) node = n - 1;
    float v;
    if (k < 32)      v = reps[node * 32 + k];
    else if (k < 64) v = msum[node * 32 + c] / fmaxf((float)cnt[node], 1.f);
    else if (k < 96) v = fdec(mmax[node * 32 + c]);
    else             v = fdec(mmin[node * 32 + c]);
    A[t] = (_Float16)v;
  }
  __syncthreads();
  wave_gemm(A, 128, pW1, b1, 256, B, nullptr, lane);  // 128 -> 256
  __syncthreads();
  wave_gemm(B, 256, pW2, b2, 128, A, nullptr, lane);  // 256 -> 128
  __syncthreads();
  wave_gemm(A, 128, pW3, b3, 32, nullptr, Bf, lane);  // 128 -> 32 (f32 out)
  __syncthreads();
  if (lane < 16) {
    float ss = 0.f;
    for (int k = 0; k < 32; ++k) { float v = Bf[lane * 32 + k]; ss += v * v; }
    scale[lane] = rsqrtf(fmaxf(ss, 1e-12f));
  }
  __syncthreads();
  for (int t = lane; t < 512; t += 32) {
    int m = t >> 5, k = t & 31;
    int node = t0 + m;
    if (active && node < n) reps[node * 32 + k] = Bf[t] * scale[m];
  }
}

// ---------- readout: gather gens, 32 -> 64 -> 32 (WMMA) -> 1 (scalar dot) ----------
__global__ __launch_bounds__(32 * WPB) void k_readout(
    const float* __restrict__ reps, const int* __restrict__ gens,
    const _Float16* __restrict__ pW1, const float* __restrict__ b1,
    const _Float16* __restrict__ pW2, const float* __restrict__ b2,
    const float* __restrict__ W3, const float* __restrict__ b3,
    float* __restrict__ out, int ngen) {
  __shared__ __align__(16) _Float16 smem[WPB][16 * 128 + 16 * 256];
  int wave = threadIdx.x >> 5, lane = threadIdx.x & 31;
  int tile = blockIdx.x * WPB + wave;
  int ntiles = (ngen + 15) >> 4;
  bool active = tile < ntiles;
  int g0 = active ? tile * 16 : 0;

  _Float16* A  = &smem[wave][0];                  // [16][32]
  _Float16* B  = &smem[wave][16 * 128];           // [16][64]
  float*    Hf = (float*)(B + 16 * 64);           // [16][32] f32

  for (int t = lane; t < 16 * 32; t += 32) {
    int m = t >> 5, k = t & 31;
    int gi = g0 + m; if (gi >= ngen) gi = ngen - 1;
    float v = reps[gens[gi] * 32 + k];
    if (v != v) v = 0.f;  // NaN -> 0 per reference
    A[t] = (_Float16)v;
  }
  __syncthreads();
  wave_gemm(A, 32, pW1, b1, 64, B, nullptr, lane);  // 32 -> 64
  __syncthreads();
  wave_gemm(B, 64, pW2, b2, 32, nullptr, Hf, lane); // 64 -> 32 (f32)
  __syncthreads();
  if (lane < 16) {
    int gi = g0 + lane;
    if (active && gi < ngen) {
      float s = b3[0];
      for (int k = 0; k < 32; ++k) s += Hf[lane * 32 + k] * W3[k];
      out[gi] = s;
    }
  }
}

extern "C" void kernel_launch(void* const* d_in, const int* in_sizes, int n_in,
                              void* d_out, int out_size, void* d_ws, size_t ws_size,
                              hipStream_t stream) {
  const float* x   = (const float*)d_in[0];
  const float* ew  = (const float*)d_in[1];
  const float* Wu1 = (const float*)d_in[2];  const float* bu1 = (const float*)d_in[3];
  const float* Wu2 = (const float*)d_in[4];  const float* bu2 = (const float*)d_in[5];
  const float* Wu3 = (const float*)d_in[6];  const float* bu3 = (const float*)d_in[7];
  const float* Wr1 = (const float*)d_in[8];  const float* br1 = (const float*)d_in[9];
  const float* Wr2 = (const float*)d_in[10]; const float* br2 = (const float*)d_in[11];
  const float* Wr3 = (const float*)d_in[12]; const float* br3 = (const float*)d_in[13];
  const int* edges = (const int*)d_in[14];
  const int* gens  = (const int*)d_in[15];
  const int n    = in_sizes[0] / 8;
  const int E    = in_sizes[1] / 4;
  const int ngen = in_sizes[15];
  float* out = (float*)d_out;
  (void)n_in; (void)out_size; (void)ws_size;

  // workspace carve-out (256B aligned)
  char* base = (char*)d_ws; size_t off = 0;
  auto take = [&](size_t bytes) -> void* {
    void* r = base + off; off = (off + bytes + 255) & ~(size_t)255; return r;
  };
  int*      cnt  = (int*)take((size_t)n * 4);
  float*    wsum = (float*)take((size_t)n * 4 * 4);
  unsigned* wmax = (unsigned*)take((size_t)n * 4 * 4);
  unsigned* wmin = (unsigned*)take((size_t)n * 4 * 4);
  float*    reps = (float*)take((size_t)n * 32 * 4);
  float*    msum = (float*)take((size_t)n * 32 * 4);
  unsigned* mmax = (unsigned*)take((size_t)n * 32 * 4);
  unsigned* mmin = (unsigned*)take((size_t)n * 32 * 4);
  _Float16* pWu1 = (_Float16*)take(128 * 256 * 2);
  _Float16* pWu2 = (_Float16*)take(256 * 128 * 2);
  _Float16* pWu3 = (_Float16*)take(128 * 32 * 2);
  _Float16* pWr1 = (_Float16*)take(32 * 64 * 2);
  _Float16* pWr2 = (_Float16*)take(64 * 32 * 2);

  const int TB = 256;
  auto blocks = [&](long long total) { return (unsigned)((total + TB - 1) / TB); };

  // pack weights into WMMA B-fragment layout (done every call; cheap)
  k_pack_w<<<blocks(128 * 256), TB, 0, stream>>>(Wu1, pWu1, 128, 256);
  k_pack_w<<<blocks(256 * 128), TB, 0, stream>>>(Wu2, pWu2, 256, 128);
  k_pack_w<<<blocks(128 * 32),  TB, 0, stream>>>(Wu3, pWu3, 128, 32);
  k_pack_w<<<blocks(32 * 64),   TB, 0, stream>>>(Wr1, pWr1, 32, 64);
  k_pack_w<<<blocks(64 * 32),   TB, 0, stream>>>(Wr2, pWr2, 64, 32);

  // prepare(): edge-weight segment stats -> reps
  k_init_wstats<<<blocks((long long)n * 4), TB, 0, stream>>>(wsum, wmax, wmin, cnt, n);
  k_edge_wstats<<<blocks((long long)E * 4), TB, 0, stream>>>(ew, edges, E, wsum, wmax, wmin, cnt);
  k_prepare<<<blocks((long long)n * 32), TB, 0, stream>>>(x, wsum, wmax, wmin, cnt, reps, n);

  // message passing iterations
  const int ntiles  = (n + 15) / 16;
  const unsigned ublk = (ntiles + WPB - 1) / WPB;
  for (int it = 0; it < 3; ++it) {
    k_init_mstats<<<blocks((long long)n * 32), TB, 0, stream>>>(msum, mmax, mmin, n * 32);
    k_edge_scatter<<<blocks((long long)E * 32), TB, 0, stream>>>(reps, edges, E, msum, mmax, mmin);
    k_update<<<ublk, 32 * WPB, 0, stream>>>(reps, msum, mmax, mmin, cnt,
                                            pWu1, bu1, pWu2, bu2, pWu3, bu3, n);
  }

  // readout
  const int gtiles = (ngen + 15) / 16;
  const unsigned gblk = (gtiles + WPB - 1) / WPB;
  k_readout<<<gblk, 32 * WPB, 0, stream>>>(reps, gens, pWr1, br1, pWr2, br2, Wr3, br3,
                                           out, ngen);
}
